// peepLSTM_3143916061189
// MI455X (gfx1250) — matrix operand order repaired and checked by reading, not compiled, not run on hardware
//
#include <hip/hip_runtime.h>
#include <hip/hip_bf16.h>
#include <math.h>

// Problem sizes (match reference)
#define B_ 128
#define T_ 256
#define I_ 512
#define H_ 1024
#define C_ 1000
#define M_ (B_ * T_)   // 32768 rows of x
#define N4_ (4 * H_)   // 4096  (f,i,o,c input projections)
#define N3_ (3 * H_)   // 3072  (f,i,o recurrent projections)

typedef __attribute__((ext_vector_type(16))) __bf16 bf16x16;
typedef __attribute__((ext_vector_type(8)))  __bf16 bf16x8;
typedef __attribute__((ext_vector_type(8)))  float  v8f;
typedef unsigned int u32x4 __attribute__((ext_vector_type(4)));
typedef int          i32x4 __attribute__((ext_vector_type(4)));
typedef int          i32x8 __attribute__((ext_vector_type(8)));

union Frag { bf16x16 v; bf16x8 h[2]; };

static __device__ __forceinline__ v8f vzero() {
  v8f z = {0.f,0.f,0.f,0.f,0.f,0.f,0.f,0.f};
  return z;
}
static __device__ __forceinline__ float sigmoidf_(float x) {
  return 1.f / (1.f + __expf(-x));
}

// ---------------------------------------------------------------------------
// Prep kernels: f32 -> bf16 conversion, weight transpose/pack, state init
// ---------------------------------------------------------------------------

__global__ void k_convert_x(const float* __restrict__ x, __bf16* __restrict__ xb, int n) {
  int i = blockIdx.x * blockDim.x + threadIdx.x;
  if (i < n) xb[i] = (__bf16)x[i];
}

// Pack W_{f,i,o,c}x [I,H] f32 into transposed bf16 W4xT [4H, I]
__global__ void k_pack_wx(const float* __restrict__ wf, const float* __restrict__ wi,
                          const float* __restrict__ wo, const float* __restrict__ wc,
                          __bf16* __restrict__ w4xt) {
  int idx = blockIdx.x * blockDim.x + threadIdx.x;   // over 4096*512
  if (idx >= N4_ * I_) return;
  int n = idx / I_;
  int k = idx % I_;
  int g = n >> 10;
  int col = n & (H_ - 1);
  const float* w = (g == 0) ? wf : (g == 1) ? wi : (g == 2) ? wo : wc;
  w4xt[(size_t)n * I_ + k] = (__bf16)w[(size_t)k * H_ + col];
}

// Pack W_{f,i,o}h [H,H] f32 into transposed bf16 W3hT [3H, H]
__global__ void k_pack_wh(const float* __restrict__ wf, const float* __restrict__ wi,
                          const float* __restrict__ wo, __bf16* __restrict__ w3ht) {
  int idx = blockIdx.x * blockDim.x + threadIdx.x;   // over 3072*1024
  if (idx >= N3_ * H_) return;
  int n = idx / H_;
  int k = idx % H_;
  int g = n >> 10;
  int col = n & (H_ - 1);
  const float* w = (g == 0) ? wf : (g == 1) ? wi : wo;
  w3ht[(size_t)n * H_ + k] = (__bf16)w[(size_t)k * H_ + col];
}

// Pack W_ph [H, C] f32 into transposed, zero-padded bf16 WpT [1024, H]
__global__ void k_pack_wp(const float* __restrict__ wp, __bf16* __restrict__ wpt) {
  int idx = blockIdx.x * blockDim.x + threadIdx.x;   // over 1024*1024
  if (idx >= H_ * H_) return;
  int n = idx / H_;
  int k = idx % H_;
  float v = (n < C_) ? wp[(size_t)k * C_ + n] : 0.f;
  wpt[(size_t)n * H_ + k] = (__bf16)v;
}

__global__ void k_init_state(__bf16* __restrict__ hb0, float* __restrict__ c) {
  int i = blockIdx.x * blockDim.x + threadIdx.x;
  if (i < B_ * H_) { hb0[i] = (__bf16)0.f; c[i] = 0.f; }
}

// ---------------------------------------------------------------------------
// Phase A: P[m][n] = sum_k xb[m][k] * W4xT[n][k]  (+gate bias), bf16 out
// Block = 256 threads (8 waves), tile 128(M) x 128(N); wave = 32x64 (2x4 wmma)
// ---------------------------------------------------------------------------
__global__ void k_gemm_x(const __bf16* __restrict__ xb,   // [M_, I_]
                         const __bf16* __restrict__ w4xt, // [N4_, I_]
                         const float* __restrict__ bfp, const float* __restrict__ bip,
                         const float* __restrict__ bop, const float* __restrict__ bcp,
                         __bf16* __restrict__ P)          // [M_, N4_]
{
  const int lane = threadIdx.x & 31;
  const int wave = threadIdx.x >> 5;
  const int wm = wave >> 1;
  const int wn = wave & 1;
  const int m0 = blockIdx.y * 128 + wm * 32;
  const int n0 = blockIdx.x * 128 + wn * 64;
  const int lrow = lane & 15;
  const int lkg  = lane >> 4;

  v8f acc[2][4];
  for (int a = 0; a < 2; ++a)
    for (int b = 0; b < 4; ++b) acc[a][b] = vzero();

  for (int k0 = 0; k0 < I_; k0 += 32) {
    Frag a[2], b[4];
    #pragma unroll
    for (int tr = 0; tr < 2; ++tr) {
      const __bf16* p = xb + (size_t)(m0 + 16 * tr + lrow) * I_ + k0 + lkg * 16;
      a[tr].h[0] = *(const bf16x8*)(p);
      a[tr].h[1] = *(const bf16x8*)(p + 8);
    }
    #pragma unroll
    for (int tc = 0; tc < 4; ++tc) {
      const __bf16* p = w4xt + (size_t)(n0 + 16 * tc + lrow) * I_ + k0 + lkg * 16;
      b[tc].h[0] = *(const bf16x8*)(p);
      b[tc].h[1] = *(const bf16x8*)(p + 8);
    }
    #pragma unroll
    for (int tr = 0; tr < 2; ++tr)
      #pragma unroll
      for (int tc = 0; tc < 4; ++tc)
        acc[tr][tc] = __builtin_amdgcn_wmma_f32_16x16x32_bf16(
            false, a[tr].v, false, b[tc].v, (short)0, acc[tr][tc], false, false);
  }

  #pragma unroll
  for (int tr = 0; tr < 2; ++tr) {
    #pragma unroll
    for (int tc = 0; tc < 4; ++tc) {
      int n = n0 + 16 * tc + lrow;
      int g = n >> 10, j = n & (H_ - 1);
      const float* bp = (g == 0) ? bfp : (g == 1) ? bip : (g == 2) ? bop : bcp;
      float bias = bp[j];
      #pragma unroll
      for (int r = 0; r < 8; ++r) {
        int m = m0 + 16 * tr + r + 8 * lkg;
        P[(size_t)m * N4_ + n] = (__bf16)(acc[tr][tc][r] + bias);
      }
    }
  }
}

// ---------------------------------------------------------------------------
// Phase B: one LSTM step.
// Block = 256 threads (8 waves), tile 32(M=batch) x 64(J=hidden cols).
// The block's h tile (32 x 1024 bf16 = 64 KB) is DMA'd into LDS once by the
// Tensor Data Mover; all waves then feed WMMA A-fragments from LDS, removing
// the 4x redundant global reads of h. B fragments stream from L2-resident
// transposed weights. Grid (H_/64, B_/32) = (16, 4).
// ---------------------------------------------------------------------------
__global__ void k_lstm_step(const __bf16* __restrict__ hb_in,  // [B_, H_]
                            __bf16* __restrict__ hb_out,       // [B_, H_]
                            float* __restrict__ c,             // [B_, H_]
                            const __bf16* __restrict__ w3ht,   // [N3_, H_]
                            const __bf16* __restrict__ P,      // [M_, N4_]
                            int t)
{
  __shared__ __bf16 atile[32 * H_];                 // 64 KB
  const int lane = threadIdx.x & 31;
  const int wave = threadIdx.x >> 5;
  const int tm = wave & 1;                          // 0..1 (m)
  const int tn = wave >> 1;                         // 0..3 (j)
  const int mblk  = blockIdx.y * 32;
  const int mbase = mblk + 16 * tm;
  const int jbase = blockIdx.x * 64 + 16 * tn;
  const int lrow = lane & 15;
  const int lkg  = lane >> 4;

  // --- TDM: DMA hb_in rows [mblk, mblk+32) -> LDS (2D tile descriptor) ---
  if (wave == 0) {
    unsigned long long ga =
        (unsigned long long)(uintptr_t)(hb_in + (size_t)mblk * H_);
    unsigned int lds = (unsigned int)(uintptr_t)(&atile[0]);  // low 32 = LDS offset
    u32x4 g0;
    g0[0] = 1u;                                        // count=1, user mode
    g0[1] = lds;                                       // lds_addr
    g0[2] = (unsigned int)(ga & 0xFFFFFFFFu);          // global_addr[31:0]
    g0[3] = (unsigned int)((ga >> 32) & 0x01FFFFFFu)   // global_addr[56:32]
            | (2u << 30);                              // type = 2 (image)
    i32x8 g1;
    g1[0] = 0x00010000;          // workgroup_mask=0, data_size=1 (2 bytes)
    g1[1] = (int)(1024u << 16);  // tensor_dim0 = 1024 (lo16)
    g1[2] = (int)(128u  << 16);  // tensor_dim0 hi=0 | tensor_dim1 = 128 (lo16)
    g1[3] = (int)(1024u << 16);  // tensor_dim1 hi=0 | tile_dim0 = 1024
    g1[4] = 32;                  // tile_dim1 = 32, tile_dim2 = 0
    g1[5] = 1024;                // tensor_dim0_stride[31:0] = 1024 elements
    g1[6] = 0;                   // stride hi | tensor_dim1_stride lo
    g1[7] = 0;
    i32x4 gz4 = {0, 0, 0, 0};                // 2D tensor: groups 2/3 unused
    i32x8 gz8 = {0, 0, 0, 0, 0, 0, 0, 0};
    __builtin_amdgcn_tensor_load_to_lds(g0, g1, gz4, gz4, gz8, 0);
    __builtin_amdgcn_s_wait_tensorcnt((short)0);
  }
  __syncthreads();

  v8f accf = vzero(), acci = vzero(), acco = vzero();
  const __bf16* arow = atile + (size_t)(16 * tm + lrow) * H_;

  for (int k0 = 0; k0 < H_; k0 += 32) {
    Frag a, bf, bi, bo;
    const __bf16* pa = arow + k0 + lkg * 16;          // ds_load_b128 x2
    a.h[0] = *(const bf16x8*)(pa);
    a.h[1] = *(const bf16x8*)(pa + 8);
    const __bf16* pb = w3ht + (size_t)(jbase + lrow) * H_ + k0 + lkg * 16;
    bf.h[0] = *(const bf16x8*)(pb);
    bf.h[1] = *(const bf16x8*)(pb + 8);
    pb += (size_t)H_ * H_;
    bi.h[0] = *(const bf16x8*)(pb);
    bi.h[1] = *(const bf16x8*)(pb + 8);
    pb += (size_t)H_ * H_;
    bo.h[0] = *(const bf16x8*)(pb);
    bo.h[1] = *(const bf16x8*)(pb + 8);

    accf = __builtin_amdgcn_wmma_f32_16x16x32_bf16(false, a.v, false, bf.v, (short)0, accf, false, false);
    acci = __builtin_amdgcn_wmma_f32_16x16x32_bf16(false, a.v, false, bi.v, (short)0, acci, false, false);
    acco = __builtin_amdgcn_wmma_f32_16x16x32_bf16(false, a.v, false, bo.v, (short)0, acco, false, false);
  }

  // Gate combine: c' = tanh(pc)*i + c*f ; h = tanh(c')*o
  const int j = jbase + lrow;
  #pragma unroll
  for (int r = 0; r < 8; ++r) {
    int m = mbase + r + 8 * lkg;                      // batch index
    size_t prow = ((size_t)m * T_ + t) * N4_;         // P row = b*T + t
    float pf = (float)P[prow + j];
    float pi = (float)P[prow + H_ + j];
    float po = (float)P[prow + 2 * H_ + j];
    float pc = (float)P[prow + 3 * H_ + j];
    float f = sigmoidf_(accf[r] + pf);
    float i = sigmoidf_(acci[r] + pi);
    float o = sigmoidf_(acco[r] + po);
    size_t ci = (size_t)m * H_ + j;
    float cn = tanhf(pc) * i + c[ci] * f;
    float hv = tanhf(cn) * o;
    c[ci] = cn;
    hb_out[ci] = (__bf16)hv;
  }
}

// ---------------------------------------------------------------------------
// Phase C: logits[m][n] = sum_k hb[m][k]*WpT[n][k] + b_p[n]  (n padded to 1024)
// ---------------------------------------------------------------------------
__global__ void k_gemm_logits(const __bf16* __restrict__ hb,   // [B_, H_]
                              const __bf16* __restrict__ wpt,  // [1024, H_]
                              const float* __restrict__ bp,    // [C_]
                              float* __restrict__ logits)      // [B_, 1024]
{
  const int lane = threadIdx.x & 31;
  const int wave = threadIdx.x >> 5;
  const int tm = wave & 1;
  const int tn = wave >> 1;
  const int mbase = blockIdx.y * 32 + 16 * tm;
  const int nbase = blockIdx.x * 64 + 16 * tn;
  const int lrow = lane & 15;
  const int lkg  = lane >> 4;

  v8f acc = vzero();
  for (int k0 = 0; k0 < H_; k0 += 32) {
    Frag a, b;
    const __bf16* pa = hb + (size_t)(mbase + lrow) * H_ + k0 + lkg * 16;
    a.h[0] = *(const bf16x8*)(pa);
    a.h[1] = *(const bf16x8*)(pa + 8);
    const __bf16* pb = wpt + (size_t)(nbase + lrow) * H_ + k0 + lkg * 16;
    b.h[0] = *(const bf16x8*)(pb);
    b.h[1] = *(const bf16x8*)(pb + 8);
    acc = __builtin_amdgcn_wmma_f32_16x16x32_bf16(false, a.v, false, b.v, (short)0, acc, false, false);
  }

  int n = nbase + lrow;
  float bias = (n < C_) ? bp[n] : 0.f;
  #pragma unroll
  for (int r = 0; r < 8; ++r) {
    int m = mbase + r + 8 * lkg;
    logits[(size_t)m * H_ + n] = acc[r] + bias;
  }
}

// One block per row; softmax over the first C_ (=1000) columns.
__global__ void k_softmax(const float* __restrict__ logits, float* __restrict__ out) {
  const int row = blockIdx.x;
  const int tid = threadIdx.x;
  __shared__ float red[256];
  const float* lr = logits + (size_t)row * H_;

  float mx = -3.402823466e38f;
  for (int j = tid; j < C_; j += 256) mx = fmaxf(mx, lr[j]);
  red[tid] = mx; __syncthreads();
  for (int s = 128; s > 0; s >>= 1) {
    if (tid < s) red[tid] = fmaxf(red[tid], red[tid + s]);
    __syncthreads();
  }
  mx = red[0]; __syncthreads();

  float sum = 0.f;
  for (int j = tid; j < C_; j += 256) sum += __expf(lr[j] - mx);
  red[tid] = sum; __syncthreads();
  for (int s = 128; s > 0; s >>= 1) {
    if (tid < s) red[tid] += red[tid + s];
    __syncthreads();
  }
  float inv = 1.f / red[0];
  for (int j = tid; j < C_; j += 256)
    out[(size_t)row * C_ + j] = __expf(lr[j] - mx) * inv;
}

// ---------------------------------------------------------------------------
// Host-side launch
// ---------------------------------------------------------------------------
extern "C" void kernel_launch(void* const* d_in, const int* in_sizes, int n_in,
                              void* d_out, int out_size, void* d_ws, size_t ws_size,
                              hipStream_t stream) {
  const float* x    = (const float*)d_in[0];
  const float* W_fx = (const float*)d_in[1];
  const float* W_fh = (const float*)d_in[2];
  const float* b_f  = (const float*)d_in[3];
  const float* W_ix = (const float*)d_in[4];
  const float* W_ih = (const float*)d_in[5];
  const float* b_i  = (const float*)d_in[6];
  const float* W_ox = (const float*)d_in[7];
  const float* W_oh = (const float*)d_in[8];
  const float* b_o  = (const float*)d_in[9];
  const float* W_cx = (const float*)d_in[10];
  const float* b_c  = (const float*)d_in[11];
  const float* W_ph = (const float*)d_in[12];
  const float* b_p  = (const float*)d_in[13];
  float* out = (float*)d_out;

  // Workspace layout (bytes) — total ~302 MB
  char* ws = (char*)d_ws;
  __bf16* xb     = (__bf16*)(ws);                         // 32 MB  [M_, I_]
  __bf16* w4xt   = (__bf16*)(ws + 33554432);              //  4 MB  [N4_, I_]
  __bf16* w3ht   = (__bf16*)(ws + 37748736);              //  6 MB  [N3_, H_]
  __bf16* wpt    = (__bf16*)(ws + 44040192);              //  2 MB  [1024, H_]
  __bf16* P      = (__bf16*)(ws + 46137344);              // 256 MB [M_, N4_]
  __bf16* hb0    = (__bf16*)(ws + 314572800);             // 256 KB
  __bf16* hb1    = (__bf16*)(ws + 314834944);             // 256 KB
  float*  cstate = (float*) (ws + 315097088);             // 512 KB
  float*  logits = (float*) (ws + 315621376);             // 512 KB

  // Prep
  k_convert_x<<<(M_ * I_ + 255) / 256, 256, 0, stream>>>(x, xb, M_ * I_);
  k_pack_wx<<<(N4_ * I_ + 255) / 256, 256, 0, stream>>>(W_fx, W_ix, W_ox, W_cx, w4xt);
  k_pack_wh<<<(N3_ * H_ + 255) / 256, 256, 0, stream>>>(W_fh, W_ih, W_oh, w3ht);
  k_pack_wp<<<(H_ * H_ + 255) / 256, 256, 0, stream>>>(W_ph, wpt);
  k_init_state<<<(B_ * H_ + 255) / 256, 256, 0, stream>>>(hb0, cstate);

  // Phase A: input projections (one big WMMA GEMM)
  k_gemm_x<<<dim3(N4_ / 128, M_ / 128), 256, 0, stream>>>(
      xb, w4xt, b_f, b_i, b_o, b_c, P);

  // Phase B: 256 sequential recurrent steps (graph-captured launches act as
  // the device-wide barrier per timestep); h double-buffered in bf16.
  for (int t = 0; t < T_; ++t) {
    const __bf16* hin = (t & 1) ? hb1 : hb0;
    __bf16* hout      = (t & 1) ? hb0 : hb1;
    k_lstm_step<<<dim3(H_ / 64, B_ / 32), 256, 0, stream>>>(
        hin, hout, cstate, w3ht, P, t);
  }
  // After t=255 (odd), final h is in hb0.

  // Phase C: projection + softmax
  k_gemm_logits<<<dim3(H_ / 64, B_ / 32), 256, 0, stream>>>(hb0, wpt, b_p, logits);
  k_softmax<<<B_, 256, 0, stream>>>(logits, out);
}
